// DualGraphSAGE_3719441678810
// MI455X (gfx1250) — matrix operand order repaired and checked by reading, not compiled, not run on hardware
//
#include <hip/hip_runtime.h>
#include <hip/hip_bf16.h>

#define NNODES 16384
#define DEG    16
#define HD     128

typedef __bf16 v16bf __attribute__((ext_vector_type(16)));
typedef __bf16 v8bf  __attribute__((ext_vector_type(8)));
typedef float  v8f   __attribute__((ext_vector_type(8)));

__device__ __forceinline__ v8f wmma_bf16f32(v16bf a, v16bf b, v8f c) {
  // D = A(16x32 bf16) * B(32x16 bf16) + C(16x16 f32), wave32 WMMA
  return __builtin_amdgcn_wmma_f32_16x16x32_bf16(false, a, false, b, (short)0, c, false, false);
}

// CDNA5 has a native v_tanh_f32 TRANS op (co-executes with WMMA). Use it for
// both tanh and sigmoid; branch-free exp/rcp fallback if the builtin is absent.
#if __has_builtin(__builtin_amdgcn_tanhf)
__device__ __forceinline__ float fast_tanh(float x) { return __builtin_amdgcn_tanhf(x); }
#elif __has_builtin(__builtin_amdgcn_tanh_f32)
__device__ __forceinline__ float fast_tanh(float x) { return __builtin_amdgcn_tanh_f32(x); }
#else
__device__ __forceinline__ float fast_tanh(float x) {
  // tanh(x) = 1 - 2/(exp(2x)+1); robust at +/-inf, no branches
  return 1.f - 2.f * __builtin_amdgcn_rcpf(__expf(2.f * x) + 1.f);
}
#endif
__device__ __forceinline__ float fast_sigmoid(float x) { return 0.5f * fast_tanh(0.5f * x) + 0.5f; }

// ---------------------------------------------------------------------------
// Pack LSTM weights into padded bf16 B-operand layout:
//   Wp[4*SP][SP]: row = padded gate-output index (gate sections strided by SP),
//   K contiguous (a B-fragment = two 16B contiguous runs per lane).
//   Padded rows/cols zero -> padded h/c columns remain exactly zero through the
//   recurrence. biasp[col] = bih+bhh (gate bias applied once per step).
// ---------------------------------------------------------------------------
__global__ void prep_weights_kernel(const float* __restrict__ wih,
                                    const float* __restrict__ whh,
                                    const float* __restrict__ bih,
                                    const float* __restrict__ bhh,
                                    int cin, int sp,
                                    __bf16* __restrict__ wihp,
                                    __bf16* __restrict__ whhp,
                                    float* __restrict__ biasp) {
  const int total = 4 * sp * sp;
  for (int i = blockIdx.x * blockDim.x + threadIdx.x; i < total; i += gridDim.x * blockDim.x) {
    const int rowp = i / sp;
    const int k    = i - rowp * sp;
    const int sec  = rowp / sp;          // gate section 0..3 (i,f,g,o)
    const int r    = rowp - sec * sp;    // row within section
    float a = 0.f, b = 0.f;
    if (r < cin && k < cin) {
      a = wih[(sec * cin + r) * cin + k];
      b = whh[(sec * cin + r) * cin + k];
    }
    wihp[i] = (__bf16)a;
    whhp[i] = (__bf16)b;
  }
  for (int i = blockIdx.x * blockDim.x + threadIdx.x; i < 4 * sp; i += gridDim.x * blockDim.x) {
    const int sec = i / sp, r = i - sec * sp;
    biasp[i] = (r < cin) ? (bih[sec * cin + r] + bhh[sec * cin + r]) : 0.f;
  }
}

// ---------------------------------------------------------------------------
// Fused gather + LSTM aggregation. One wave owns 16 nodes (one WMMA M-tile).
// Per step t: gates[16 x 4*SP] = x_t @ Wih^T + h @ Whh^T + bias via
// v_wmma_f32_16x16x32_bf16 (f32 accumulate). Cell state c stays in f32 C-tiles
// in registers; h round-trips through a per-wave LDS scratch (bf16) to convert
// C/D layout -> A-fragment layout for the next step.
// Weights are LDS-resident (dynamic LDS: 280.5KB for CIN=128 < 320KB/WGP), so
// steady-state B traffic never leaves the WGP; only the L2-resident x gathers
// use VMEM. Dynamic LDS carve: [sWih | sWhh | sBias | sH(per wave)].
// ---------------------------------------------------------------------------
template <int CIN, int SP, int WAVES>
__global__ __launch_bounds__(WAVES * 32) void lstm_agg_kernel(
    const float* __restrict__ xin,      // [N][CIN] f32 node features
    const int* __restrict__ esrc,       // [N*DEG] neighbor sources (sorted by dst)
    const __bf16* __restrict__ Wih,     // [4*SP][SP] padded bf16
    const __bf16* __restrict__ Whh,     // [4*SP][SP] padded bf16
    const float* __restrict__ biasp,    // [4*SP]
    float* __restrict__ agg)            // [N][SP] f32 out (cols >= CIN are 0)
{
  constexpr int NKF = SP / 32;          // K fragments of 32
  constexpr int CT  = SP / 16;          // 16-wide c-column tiles
  constexpr int WELEMS = 4 * SP * SP;   // elements per weight matrix

  extern __shared__ __align__(16) char dynsmem[];
  __bf16* sWih  = (__bf16*)dynsmem;
  __bf16* sWhh  = sWih + WELEMS;
  float*  sBias = (float*)(sWhh + WELEMS);
  __bf16* sHall = (__bf16*)(sBias + 4 * SP);

  const int tid  = threadIdx.x;
  const int wv   = tid >> 5;
  const int lane = tid & 31;
  const int row  = lane & 15;           // M row (node within tile) / N column
  const int kh   = lane >> 4;           // K-half selector for A/B fragments
  __bf16* sH = sHall + wv * (16 * SP);

  // cooperative copy: global (prepped bf16) -> LDS, 16B vectors
  {
    const int4* gi = (const int4*)Wih;
    const int4* gh = (const int4*)Whh;
    int4* li = (int4*)sWih;
    int4* lh = (int4*)sWhh;
    for (int i = tid; i < WELEMS / 8; i += WAVES * 32) { li[i] = gi[i]; lh[i] = gh[i]; }
  }
  for (int i = tid; i < 4 * SP; i += WAVES * 32) sBias[i] = biasp[i];
  for (int i = lane; i < 16 * SP; i += 32) sH[i] = (__bf16)0.f;  // h0 = 0
  __syncthreads();

  const int nodeBase = (blockIdx.x * WAVES + wv) * 16;
  const int node     = nodeBase + row;

  v8f c[CT];                            // cell state, f32 C-tile layout
#pragma unroll
  for (int j = 0; j < CT; ++j)
#pragma unroll
    for (int r = 0; r < 8; ++r) c[j][r] = 0.f;

#pragma unroll 1
  for (int t = 0; t < DEG; ++t) {
    // ---- A-fragments for x_t: fused neighbor gather from L2-resident x ----
    const int s = esrc[node * DEG + t];
    const float* xrow = xin + (long long)s * CIN;
    v16bf ax[NKF];
#pragma unroll
    for (int kf = 0; kf < NKF; ++kf) {
      const int k0 = kf * 32 + kh * 8;       // vector elems 0..7  -> these K
      const int k1 = kf * 32 + 16 + kh * 8;  // vector elems 8..15 -> these K
      if constexpr (CIN % 32 == 0) {
        v8f lo = *(const v8f*)(xrow + k0);
        v8f hi = *(const v8f*)(xrow + k1);
#pragma unroll
        for (int e = 0; e < 8; ++e) { ax[kf][e] = (__bf16)lo[e]; ax[kf][e + 8] = (__bf16)hi[e]; }
      } else {
#pragma unroll
        for (int e = 0; e < 8; ++e) {
          const int ka = k0 + e, kb = k1 + e;
          ax[kf][e]     = (ka < CIN) ? (__bf16)xrow[ka] : (__bf16)0.f;
          ax[kf][e + 8] = (kb < CIN) ? (__bf16)xrow[kb] : (__bf16)0.f;
        }
      }
    }
    // ---- A-fragments for h from per-wave LDS scratch ----
    v16bf ah[NKF];
    const __bf16* hrow = sH + row * SP;
#pragma unroll
    for (int kf = 0; kf < NKF; ++kf) {
      v8bf lo = *(const v8bf*)(hrow + kf * 32 + kh * 8);
      v8bf hi = *(const v8bf*)(hrow + kf * 32 + 16 + kh * 8);
#pragma unroll
      for (int e = 0; e < 8; ++e) { ah[kf][e] = lo[e]; ah[kf][e + 8] = hi[e]; }
    }

    // ---- gate GEMMs + pointwise LSTM update, one 16-col c-tile at a time ----
#pragma unroll 1
    for (int j = 0; j < CT; ++j) {
      v8f gacc[4];
#pragma unroll
      for (int g2 = 0; g2 < 4; ++g2) {
        v8f acc;
#pragma unroll
        for (int r = 0; r < 8; ++r) acc[r] = 0.f;
        const int col = g2 * SP + j * 16 + row;  // lane's B column
        const __bf16* wi = sWih + col * SP + kh * 8;
        const __bf16* wh = sWhh + col * SP + kh * 8;
#pragma unroll
        for (int kf = 0; kf < NKF; ++kf) {
          v8bf blo = *(const v8bf*)(wi + kf * 32);
          v8bf bhi = *(const v8bf*)(wi + kf * 32 + 16);
          v16bf b;
#pragma unroll
          for (int e = 0; e < 8; ++e) { b[e] = blo[e]; b[e + 8] = bhi[e]; }
          acc = wmma_bf16f32(ax[kf], b, acc);
        }
#pragma unroll
        for (int kf = 0; kf < NKF; ++kf) {
          v8bf blo = *(const v8bf*)(wh + kf * 32);
          v8bf bhi = *(const v8bf*)(wh + kf * 32 + 16);
          v16bf b;
#pragma unroll
          for (int e = 0; e < 8; ++e) { b[e] = blo[e]; b[e + 8] = bhi[e]; }
          acc = wmma_bf16f32(ah[kf], b, acc);
        }
        gacc[g2] = acc;
      }
      const float bi  = sBias[0 * SP + j * 16 + row];
      const float bff = sBias[1 * SP + j * 16 + row];
      const float bg  = sBias[2 * SP + j * 16 + row];
      const float bo  = sBias[3 * SP + j * 16 + row];
      __bf16* hdst = sH + (kh * 8) * SP + j * 16 + row;
#pragma unroll
      for (int r = 0; r < 8; ++r) {
        const float iv = fast_sigmoid(gacc[0][r] + bi);
        const float fv = fast_sigmoid(gacc[1][r] + bff);
        const float gv = fast_tanh(gacc[2][r] + bg);
        const float ov = fast_sigmoid(gacc[3][r] + bo);
        const float cv = fv * c[j][r] + iv * gv;
        c[j][r] = cv;
        hdst[r * SP] = (__bf16)(ov * fast_tanh(cv));  // C-layout elem (M=r+kh*8, col)
      }
    }
    __syncthreads();
  }

  // write last hidden state as f32 aggregation
  for (int i = lane; i < 16 * SP; i += 32) {
    const int m = i / SP, cc = i - m * SP;
    agg[(long long)(nodeBase + m) * SP + cc] = (float)sH[i];
  }
}

// ---------------------------------------------------------------------------
// h = agg @ wl^T + bl + x @ wr^T ; LayerNorm ; ReLU ; optional residual.
// One 128-thread block per node.
// ---------------------------------------------------------------------------
__global__ __launch_bounds__(HD) void sage_ln_kernel(
    const float* __restrict__ xin, int cin, int aggStride,
    const float* __restrict__ agg,
    const float* __restrict__ wl, const float* __restrict__ bl,
    const float* __restrict__ wr,
    const float* __restrict__ gam, const float* __restrict__ bet,
    const float* __restrict__ xres, int doRes,
    float* __restrict__ xout)
{
  const int n = blockIdx.x;
  const int cthr = threadIdx.x;
  const float* arow = agg + (long long)n * aggStride;
  const float* xrow = xin + (long long)n * cin;
  const float* wlr  = wl + (long long)cthr * cin;
  const float* wrr  = wr + (long long)cthr * cin;
  float acc = bl[cthr];
  for (int k = 0; k < cin; ++k) acc += arow[k] * wlr[k] + xrow[k] * wrr[k];

  __shared__ float red[HD];
  red[cthr] = acc;
  __syncthreads();
  for (int s2 = HD / 2; s2 > 0; s2 >>= 1) { if (cthr < s2) red[cthr] += red[cthr + s2]; __syncthreads(); }
  const float mu = red[0] * (1.f / HD);
  __syncthreads();
  const float d = acc - mu;
  red[cthr] = d * d;
  __syncthreads();
  for (int s2 = HD / 2; s2 > 0; s2 >>= 1) { if (cthr < s2) red[cthr] += red[cthr + s2]; __syncthreads(); }
  const float var = red[0] * (1.f / HD);

  float y = d * rsqrtf(var + 1e-5f) * gam[cthr] + bet[cthr];
  y = fmaxf(y, 0.f);
  if (doRes) y += xres[(long long)n * HD + cthr];
  xout[(long long)n * HD + cthr] = y;
}

// ---------------------------------------------------------------------------
// out[n] = relu(x[n] @ cw1^T + cb1) @ cw2^T + cb2. 64 threads per node.
// ---------------------------------------------------------------------------
__global__ __launch_bounds__(64) void mlp_head_kernel(
    const float* __restrict__ x,
    const float* __restrict__ cw1, const float* __restrict__ cb1,
    const float* __restrict__ cw2, const float* __restrict__ cb2,
    float* __restrict__ out)
{
  const int n = blockIdx.x, j = threadIdx.x;
  const float* xr = x + (long long)n * HD;
  const float* w  = cw1 + (long long)j * HD;
  float acc = cb1[j];
  for (int k = 0; k < HD; ++k) acc += xr[k] * w[k];
  acc = fmaxf(acc, 0.f) * cw2[j];
  __shared__ float red[64];
  red[j] = acc;
  __syncthreads();
  for (int s2 = 32; s2 > 0; s2 >>= 1) { if (j < s2) red[j] += red[j + s2]; __syncthreads(); }
  if (j == 0) out[n] = red[0] + cb2[0];
}

extern "C" void kernel_launch(void* const* d_in, const int* in_sizes, int n_in,
                              void* d_out, int out_size, void* d_ws, size_t ws_size,
                              hipStream_t stream) {
  (void)in_sizes; (void)n_in; (void)out_size; (void)ws_size;
  const float* x0 = (const float*)d_in[0];
  auto L = [&](int l, int off) { return (const float*)d_in[1 + 9 * l + off]; };
  const float* cw1 = (const float*)d_in[28];
  const float* cb1 = (const float*)d_in[29];
  const float* cw2 = (const float*)d_in[30];
  const float* cb2 = (const float*)d_in[31];
  const int* esrc  = (const int*)d_in[32];
  // d_in[33] = edge_dst: implied by fixed-degree sorted layout, unused.

  char* w = (char*)d_ws;
  float* xbuf0 = (float*)(w);
  float* xbuf1 = (float*)(w + (size_t)8 * 1024 * 1024);
  float* agg   = (float*)(w + (size_t)16 * 1024 * 1024);
  char* wp = w + (size_t)24 * 1024 * 1024;
  __bf16* wihp0 = (__bf16*)wp; wp += 4 * 32 * 32 * 2;
  __bf16* whhp0 = (__bf16*)wp; wp += 4 * 32 * 32 * 2;
  float*  biasp0 = (float*)wp; wp += 4 * 32 * 4;
  __bf16* wihp1 = (__bf16*)wp; wp += 4 * 128 * 128 * 2;
  __bf16* whhp1 = (__bf16*)wp; wp += 4 * 128 * 128 * 2;
  float*  biasp1 = (float*)wp; wp += 4 * 128 * 4;
  __bf16* wihp2 = (__bf16*)wp; wp += 4 * 128 * 128 * 2;
  __bf16* whhp2 = (__bf16*)wp; wp += 4 * 128 * 128 * 2;
  float*  biasp2 = (float*)wp; wp += 4 * 128 * 4;

  prep_weights_kernel<<<16, 256, 0, stream>>>(L(0,0), L(0,1), L(0,2), L(0,3), 18, 32, wihp0, whhp0, biasp0);
  prep_weights_kernel<<<256, 256, 0, stream>>>(L(1,0), L(1,1), L(1,2), L(1,3), 128, 128, wihp1, whhp1, biasp1);
  prep_weights_kernel<<<256, 256, 0, stream>>>(L(2,0), L(2,1), L(2,2), L(2,3), 128, 128, wihp2, whhp2, biasp2);

  // dynamic LDS: [Wih | Whh | bias | h-scratch], per-workgroup
  const size_t smem0 = (size_t)2 * (4 * 32 * 32 * 2) + 4 * 32 * 4 + 4 * 16 * 32 * 2;      // 20,992 B
  const size_t smem1 = (size_t)2 * (4 * 128 * 128 * 2) + 4 * 128 * 4 + 4 * 16 * 128 * 2;  // 280,576 B

  // layer 0: Cin=18 (padded to 32), no residual
  lstm_agg_kernel<18, 32, 4><<<NNODES / (16 * 4), 4 * 32, smem0, stream>>>(x0, esrc, wihp0, whhp0, biasp0, agg);
  sage_ln_kernel<<<NNODES, HD, 0, stream>>>(x0, 18, 32, agg, L(0,4), L(0,5), L(0,6), L(0,7), L(0,8), nullptr, 0, xbuf0);
  // layer 1: Cin=128, residual
  lstm_agg_kernel<128, 128, 4><<<NNODES / (16 * 4), 4 * 32, smem1, stream>>>(xbuf0, esrc, wihp1, whhp1, biasp1, agg);
  sage_ln_kernel<<<NNODES, HD, 0, stream>>>(xbuf0, 128, 128, agg, L(1,4), L(1,5), L(1,6), L(1,7), L(1,8), xbuf0, 1, xbuf1);
  // layer 2: Cin=128, residual
  lstm_agg_kernel<128, 128, 4><<<NNODES / (16 * 4), 4 * 32, smem1, stream>>>(xbuf1, esrc, wihp2, whhp2, biasp2, agg);
  sage_ln_kernel<<<NNODES, HD, 0, stream>>>(xbuf1, 128, 128, agg, L(2,4), L(2,5), L(2,6), L(2,7), L(2,8), xbuf1, 1, xbuf0);
  // classifier head
  mlp_head_kernel<<<NNODES, 64, 0, stream>>>(xbuf0, cw1, cb1, cw2, cb2, (float*)d_out);
}
